// TriangleMultiplicativeUpdate_82308753260998
// MI455X (gfx1250) — compile-verified, matrix operand order checked by reading, not run on hardware
//
#include <hip/hip_runtime.h>
#include <hip/hip_bf16.h>

typedef _Float16 half_t;
typedef __attribute__((ext_vector_type(16))) _Float16 v16h;
typedef __attribute__((ext_vector_type(8)))  _Float16 v8h;
typedef __attribute__((ext_vector_type(8)))  float    v8f;

#define N_SEQ 512
#define CDIM  128
#define NPAIR (N_SEQ * N_SEQ)   // 262144
#define LN_EPS 1e-5f

__device__ __forceinline__ float sigmoidf(float x) {
    return 1.0f / (1.0f + __expf(-x));
}

// Build a 16x32 f16 WMMA fragment (A or B): two contiguous 8-half (16B) chunks.
// lane l (l%16 = row/col): chunk0 = K[base + 8*hi .. +7], chunk1 = K[base+16+8*hi .. +7]
__device__ __forceinline__ v16h load_frag16(const half_t* c0, const half_t* c1) {
    v16h r;
    *(v8h*)&r       = *(const v8h*)c0;
    *((v8h*)&r + 1) = *(const v8h*)c1;
    return r;
}

// LDS byte offset of a __shared__ object: flat shared addr = {aperture_hi, lds_off}
__device__ __forceinline__ uint32_t lds_off(const void* p) {
    return (uint32_t)(uintptr_t)p;
}

// CDNA5 async copy: LDS[lds_addr] <= MEM[gaddr], 16 bytes per lane, ASYNCcnt-tracked.
__device__ __forceinline__ void async_load_b128(uint32_t lds_addr, const void* gaddr) {
    asm volatile("global_load_async_to_lds_b128 %0, %1, off"
                 :: "v"(lds_addr), "v"(gaddr)
                 : "memory");
}

// ---------------------------------------------------------------------------
// Kernel 0: convert the six 128x128 f32 weight matrices to f16 in workspace.
// slot order: 0=ag_w 1=ap_w 2=bg_w 3=bp_w 4=g_w 5=z_w
// ---------------------------------------------------------------------------
__global__ void wconv_kernel(const float* __restrict__ ag, const float* __restrict__ ap,
                             const float* __restrict__ bg, const float* __restrict__ bp,
                             const float* __restrict__ gw, const float* __restrict__ zw,
                             half_t* __restrict__ w16) {
    int idx = blockIdx.x * blockDim.x + threadIdx.x;   // 0 .. 6*16384-1
    int mat = idx >> 14;
    int off = idx & 16383;
    const float* src;
    switch (mat) {
        case 0:  src = ag; break;
        case 1:  src = ap; break;
        case 2:  src = bg; break;
        case 3:  src = bp; break;
        case 4:  src = gw; break;
        default: src = zw; break;
    }
    w16[idx] = (half_t)src[off];
}

// ---------------------------------------------------------------------------
// Kernel 1: LayerNorm(z) + 5 gated projections via WMMA.
//   outputs: wA  [d][i*512+k]  f16 (channel-major, row-major i x k)
//            wBT [d][j*512+k]  f16 (channel-major, TRANSPOSED: j x k)
//            wG  [d][r]        f16 (channel-major)
// One block = 16 rows of the flattened [262144, 128] problem, 4 waves.
// ---------------------------------------------------------------------------
__global__ __launch_bounds__(128)
void k1_ln_proj(const float* __restrict__ z, const float* __restrict__ mask,
                const float* __restrict__ lnw, const float* __restrict__ lnb,
                const half_t* __restrict__ w16,
                const float* __restrict__ ag_b, const float* __restrict__ ap_b,
                const float* __restrict__ bg_b, const float* __restrict__ bp_b,
                const float* __restrict__ g_b,
                half_t* __restrict__ wA, half_t* __restrict__ wBT,
                half_t* __restrict__ wG)
{
    __shared__ float  zraw[16][CDIM];
    __shared__ float  mval[16];
    __shared__ float  mu[16], rs[16];
    __shared__ half_t znh[16][136];        // padded stride: 136 halves = 68 dwords

    const int t  = threadIdx.x;
    const int rb = blockIdx.x * 16;

    // ---- load 16x128 z tile (each thread: 16 consecutive f32) ----
    {
        int rr = t >> 3;
        int c0 = (t & 7) * 16;
        const float* src = z + (size_t)(rb + rr) * CDIM + c0;
        #pragma unroll
        for (int c = 0; c < 16; ++c) zraw[rr][c0 + c] = src[c];
        if (t < 16) mval[t] = mask[rb + t];
    }
    __syncthreads();
    // ---- per-row mean / rstd ----
    if (t < 16) {
        float s = 0.f;
        #pragma unroll 8
        for (int c = 0; c < CDIM; ++c) s += zraw[t][c];
        float m = s * (1.0f / CDIM);
        float v = 0.f;
        #pragma unroll 8
        for (int c = 0; c < CDIM; ++c) { float d = zraw[t][c] - m; v += d * d; }
        mu[t] = m;
        rs[t] = rsqrtf(v * (1.0f / CDIM) + LN_EPS);
    }
    __syncthreads();
    // ---- normalize -> f16 LDS ----
    {
        int rr = t >> 3;
        int c0 = (t & 7) * 16;
        float m = mu[rr], r = rs[rr];
        #pragma unroll
        for (int c = 0; c < 16; ++c) {
            int cc = c0 + c;
            znh[rr][cc] = (half_t)((zraw[rr][cc] - m) * r * lnw[cc] + lnb[cc]);
        }
    }
    __syncthreads();

    const int lane = t & 31;
    const int wid  = t >> 5;
    const int lm   = lane & 15;
    const int hi   = lane >> 4;

    // ---- A fragments (shared across all 5 matrices), 4 K-steps of 32 ----
    v16h afrag[4];
    #pragma unroll
    for (int ks = 0; ks < 4; ++ks) {
        const half_t* p = &znh[lm][ks * 32 + 8 * hi];
        afrag[ks] = load_frag16(p, p + 16);
    }

    // each wave: 2 of the 8 N-tiles
    for (int nt = wid; nt < 8; nt += 4) {
        const int nb = nt * 16;
        const int d  = nb + lm;         // output channel owned by this lane

        auto mm = [&](const half_t* W) -> v8f {
            v8f acc = {};
            #pragma unroll
            for (int ks = 0; ks < 4; ++ks) {
                const half_t* p = W + (size_t)d * CDIM + ks * 32 + 8 * hi;
                v16h bfrag = load_frag16(p, p + 16);
                acc = __builtin_amdgcn_wmma_f32_16x16x32_f16(
                          false, afrag[ks], false, bfrag, (short)0, acc, false, false);
            }
            return acc;
        };

        // ---- a = mask * sigmoid(zn@ag^T + b) * (zn@ap^T + b) ----
        {
            v8f accg = mm(w16 + 0 * 16384);
            v8f accp = mm(w16 + 1 * 16384);
            float bg_ = ag_b[d], bp_ = ap_b[d];
            v8h pack;
            #pragma unroll
            for (int vr = 0; vr < 8; ++vr) {
                int m = vr + 8 * hi;
                pack[vr] = (half_t)(mval[m] * sigmoidf(accg[vr] + bg_) * (accp[vr] + bp_));
            }
            *(v8h*)(wA + (size_t)d * NPAIR + rb + 8 * hi) = pack;   // 16B contiguous
        }

        // ---- b (stored transposed: wBT[d][j][k_tri]) ----
        {
            v8f accg = mm(w16 + 2 * 16384);
            v8f accp = mm(w16 + 3 * 16384);
            float bg_ = bg_b[d], bp_ = bp_b[d];
            int r0   = rb + 8 * hi;           // 8 consecutive rows, same k_tri
            int ktri = r0 >> 9;
            int j0   = r0 & 511;
            half_t* dst = wBT + (size_t)d * NPAIR + (size_t)j0 * N_SEQ + ktri;
            #pragma unroll
            for (int vr = 0; vr < 8; ++vr) {
                float v = mval[vr + 8 * hi] * sigmoidf(accg[vr] + bg_) * (accp[vr] + bp_);
                dst[(size_t)vr * N_SEQ] = (half_t)v;
            }
        }

        // ---- g = sigmoid(zn@g^T + b) ----
        {
            v8f accG = mm(w16 + 4 * 16384);
            float gb_ = g_b[d];
            v8h pack;
            #pragma unroll
            for (int vr = 0; vr < 8; ++vr)
                pack[vr] = (half_t)sigmoidf(accG[vr] + gb_);
            *(v8h*)(wG + (size_t)d * NPAIR + rb + 8 * hi) = pack;
        }
    }
}

// ---------------------------------------------------------------------------
// Kernel 2: triangle einsum = 128 independent 512x512x512 f16 GEMMs.
// grid (8, 4, 128): block computes a 128(M) x 64(N) tile of X_d = A_d * B_d.
// 4 waves; each wave owns a 32x64 strip => 8 WMMAs per k-step.
// Tiles stream into double-buffered LDS via CDNA5 async loads (ASYNCcnt).
// Output wX channel-major f32: wX[d][i*512+j].
// ---------------------------------------------------------------------------
#define K2PAD 40
__global__ __launch_bounds__(128)
void k2_triangle(const half_t* __restrict__ wA, const half_t* __restrict__ wBT,
                 float* __restrict__ wX)
{
    __shared__ half_t As[2][128][K2PAD];   // 2 x 10240 B
    __shared__ half_t Bs[2][64][K2PAD];    // 2 x  5120 B

    const int d  = blockIdx.z;
    const int ib = blockIdx.y;             // 0..3  (128-row strip)
    const int jb = blockIdx.x;             // 0..7  (64-col strip)
    const int t  = threadIdx.x;
    const int lane = t & 31, wid = t >> 5;
    const int lm = lane & 15, hi = lane >> 4;

    const half_t* Ad = wA  + (size_t)d * NPAIR + (size_t)(ib * 128) * N_SEQ;
    const half_t* Bd = wBT + (size_t)d * NPAIR + (size_t)(jb * 64)  * N_SEQ;

    // issue 6 async b128 loads per thread: A tile 128x32 (512 chunks),
    // B tile 64x32 (256 chunks); chunk = 8 halves (16B).
    auto issue = [&](int bu, int kb) {
        #pragma unroll
        for (int p = 0; p < 4; ++p) {
            int q   = t + 128 * p;         // 0..511
            int row = q >> 2;
            int co  = (q & 3) * 8;
            async_load_b128(lds_off(&As[bu][row][co]),
                            Ad + (size_t)row * N_SEQ + kb + co);
        }
        #pragma unroll
        for (int p = 0; p < 2; ++p) {
            int q   = t + 128 * p;         // 0..255
            int row = q >> 2;
            int co  = (q & 3) * 8;
            async_load_b128(lds_off(&Bs[bu][row][co]),
                            Bd + (size_t)row * N_SEQ + kb + co);
        }
    };

    v8f acc[2][4] = {};

    issue(0, 0);
    for (int s = 0; s < 16; ++s) {
        const int bu = s & 1;
        if (s + 1 < 16) {
            issue(bu ^ 1, (s + 1) * 32);   // prefetch next buffer
            asm volatile("s_wait_asynccnt 6" ::: "memory");  // current buffer done
        } else {
            asm volatile("s_wait_asynccnt 0" ::: "memory");
        }
        __syncthreads();                   // all waves' tile data visible

        v16h af[2];
        #pragma unroll
        for (int mt = 0; mt < 2; ++mt) {
            const half_t* pa = &As[bu][wid * 32 + mt * 16 + lm][8 * hi];
            af[mt] = load_frag16(pa, pa + 16);
        }
        #pragma unroll
        for (int jt = 0; jt < 4; ++jt) {
            const half_t* pb = &Bs[bu][jt * 16 + lm][8 * hi];
            v16h bf = load_frag16(pb, pb + 16);
            #pragma unroll
            for (int mt = 0; mt < 2; ++mt)
                acc[mt][jt] = __builtin_amdgcn_wmma_f32_16x16x32_f16(
                                  false, af[mt], false, bf, (short)0,
                                  acc[mt][jt], false, false);
        }
        __syncthreads();                   // done reading bu before it is re-filled
    }

    float* Xd = wX + (size_t)d * NPAIR;
    #pragma unroll
    for (int mt = 0; mt < 2; ++mt) {
        #pragma unroll
        for (int jt = 0; jt < 4; ++jt) {
            int j = jb * 64 + jt * 16 + lm;
            #pragma unroll
            for (int vr = 0; vr < 8; ++vr) {
                int i = ib * 128 + wid * 32 + mt * 16 + vr + 8 * hi;
                Xd[(size_t)i * N_SEQ + j] = acc[mt][jt][vr];  // lanes 0..15: 64B coalesced
            }
        }
    }
}

// ---------------------------------------------------------------------------
// Kernel 3: gather x, LayerNorm over C, project with z_w (WMMA), gate, store.
// One block = 16 output rows.
// ---------------------------------------------------------------------------
__global__ __launch_bounds__(128)
void k3_out(const float* __restrict__ wX, const half_t* __restrict__ w16,
            const float* __restrict__ lnw, const float* __restrict__ lnb,
            const float* __restrict__ zb, const half_t* __restrict__ wG,
            float* __restrict__ out)
{
    __shared__ float  xt[16][132];     // padded stride 132 dwords
    __shared__ float  mu[16], rs[16];
    __shared__ half_t xnh[16][136];

    const int t  = threadIdx.x;
    const int rb = blockIdx.x * 16;
    const int rr = t & 15;
    const int dg = t >> 4;             // 0..7

    // gather x from channel-major: per 16-thread group, 64B coalesced
    #pragma unroll
    for (int it = 0; it < 16; ++it) {
        int d = dg + 8 * it;
        xt[rr][d] = wX[(size_t)d * NPAIR + rb + rr];
    }
    __syncthreads();
    if (t < 16) {
        float s = 0.f;
        #pragma unroll 8
        for (int c = 0; c < CDIM; ++c) s += xt[t][c];
        float m = s * (1.0f / CDIM);
        float v = 0.f;
        #pragma unroll 8
        for (int c = 0; c < CDIM; ++c) { float dd = xt[t][c] - m; v += dd * dd; }
        mu[t] = m;
        rs[t] = rsqrtf(v * (1.0f / CDIM) + LN_EPS);
    }
    __syncthreads();
    {
        float m = mu[rr], r = rs[rr];
        #pragma unroll
        for (int q = 0; q < 16; ++q) {
            int c = dg * 16 + q;
            xnh[rr][c] = (half_t)((xt[rr][c] - m) * r * lnw[c] + lnb[c]);
        }
    }
    __syncthreads();

    const int lane = t & 31, wid = t >> 5;
    const int lm = lane & 15, hi = lane >> 4;

    v16h afrag[4];
    #pragma unroll
    for (int ks = 0; ks < 4; ++ks) {
        const half_t* p = &xnh[lm][ks * 32 + 8 * hi];
        afrag[ks] = load_frag16(p, p + 16);
    }

    const half_t* Wz = w16 + 5 * 16384;
    for (int nt = wid; nt < 8; nt += 4) {
        const int nb  = nt * 16;
        const int dch = nb + lm;
        v8f acc = {};
        #pragma unroll
        for (int ks = 0; ks < 4; ++ks) {
            const half_t* p = Wz + (size_t)dch * CDIM + ks * 32 + 8 * hi;
            v16h bfrag = load_frag16(p, p + 16);
            acc = __builtin_amdgcn_wmma_f32_16x16x32_f16(
                      false, afrag[ks], false, bfrag, (short)0, acc, false, false);
        }
        float bias = zb[dch];
        v8h gv = *(const v8h*)(wG + (size_t)dch * NPAIR + rb + 8 * hi);
        #pragma unroll
        for (int vr = 0; vr < 8; ++vr) {
            int m = vr + 8 * hi;
            out[(size_t)(rb + m) * CDIM + dch] = (acc[vr] + bias) * (float)gv[vr];
        }
    }
}

// ---------------------------------------------------------------------------
// Workspace layout (bytes):
//   [0,           67108864)  wA   f16 [128][262144]
//   [67108864,   134217728)  wBT  f16 [128][512][512] (j-major)
//   [134217728,  201326592)  wG   f16 [128][262144]
//   [201326592,  335544320)  wX   f32 [128][262144]
//   [335544320,  ...      )  w16  f16 6 x 128x128 weights
// ---------------------------------------------------------------------------
extern "C" void kernel_launch(void* const* d_in, const int* in_sizes, int n_in,
                              void* d_out, int out_size, void* d_ws, size_t ws_size,
                              hipStream_t stream)
{
    const float* z        = (const float*)d_in[0];
    const float* mask     = (const float*)d_in[1];
    const float* ln_in_w  = (const float*)d_in[2];
    const float* ln_in_b  = (const float*)d_in[3];
    const float* ag_w     = (const float*)d_in[4];
    const float* ag_b     = (const float*)d_in[5];
    const float* ap_w     = (const float*)d_in[6];
    const float* ap_b     = (const float*)d_in[7];
    const float* bg_w     = (const float*)d_in[8];
    const float* bg_b     = (const float*)d_in[9];
    const float* bp_w     = (const float*)d_in[10];
    const float* bp_b     = (const float*)d_in[11];
    const float* ln_out_w = (const float*)d_in[12];
    const float* ln_out_b = (const float*)d_in[13];
    const float* z_w      = (const float*)d_in[14];
    const float* z_b      = (const float*)d_in[15];
    const float* g_w      = (const float*)d_in[16];
    const float* g_b      = (const float*)d_in[17];

    char* ws = (char*)d_ws;
    half_t* wA  = (half_t*)(ws);
    half_t* wBT = (half_t*)(ws + (size_t)67108864);
    half_t* wG  = (half_t*)(ws + (size_t)134217728);
    float*  wX  = (float* )(ws + (size_t)201326592);
    half_t* w16 = (half_t*)(ws + (size_t)335544320);

    wconv_kernel<<<384, 256, 0, stream>>>(ag_w, ap_w, bg_w, bp_w, g_w, z_w, w16);

    k1_ln_proj<<<NPAIR / 16, 128, 0, stream>>>(z, mask, ln_in_w, ln_in_b, w16,
                                               ag_b, ap_b, bg_b, bp_b, g_b,
                                               wA, wBT, wG);

    dim3 g2(8, 4, 128);
    k2_triangle<<<g2, 128, 0, stream>>>(wA, wBT, wX);

    k3_out<<<NPAIR / 16, 128, 0, stream>>>(wX, w16, ln_out_w, ln_out_b, z_b, wG,
                                           (float*)d_out);
}
